// WMEGNN_22316650070614
// MI455X (gfx1250) — matrix operand order repaired and checked by reading, not compiled, not run on hardware
//
#include <hip/hip_runtime.h>
#include <math.h>

// ---------------------------------------------------------------------------
// MI455X (gfx1250) fused EGNN. wave32, WMMA bf16 16x16x32, 2x2 register tiles.
// Edge MLP decomposed: edg1(concat(hi,hj,d2)) = Aterm[i] + Bterm[j] + d2*w1c.
// Per-edge edg2/coord1 run as bf16 WMMA on 128-edge tiles resident in LDS.
// Weights pre-transposed [N][K] bf16 -> contiguous B-fragments (ds_load_b128).
// Weight panels staged with global_load_async_to_lds_b128 + s_wait_asynccnt.
// ---------------------------------------------------------------------------

typedef __attribute__((ext_vector_type(16))) __bf16 v16bf;
typedef __attribute__((ext_vector_type(8)))  __bf16 v8bf;
typedef __attribute__((ext_vector_type(4)))  __bf16 v4bf;
typedef __attribute__((ext_vector_type(8)))  float  v8f;

#define NN   256
#define BB   4

__device__ __forceinline__ float fast_rcp(float x) {
#if __has_builtin(__builtin_amdgcn_rcpf)
  return __builtin_amdgcn_rcpf(x);
#else
  return 1.f / x;
#endif
}
__device__ __forceinline__ float sigm(float v) { return fast_rcp(1.f + __expf(-v)); }
__device__ __forceinline__ float fast_tanh(float x) {
#if __has_builtin(__builtin_amdgcn_tanhf)
  return __builtin_amdgcn_tanhf(x);
#else
  float e = __expf(2.f * x);
  return 1.f - 2.f * fast_rcp(e + 1.f);
#endif
}

__device__ __forceinline__ v8f wmma_bf16(v16bf a, v16bf b, v8f c) {
  return __builtin_amdgcn_wmma_f32_16x16x32_bf16(false, a, false, b,
                                                 (short)0, c, false, false);
}

// A fragment: 16x32 bf16 tile, row-major, ld elems/row.  Two 16B LDS loads.
__device__ __forceinline__ v16bf load_a_frag(const __bf16* base, int ld, int lane) {
  int m = lane & 15, kb = (lane >> 4) << 4;
  const __bf16* r = base + m * ld + kb;
  union { v16bf v; v8bf h[2]; } u;
  u.h[0] = *(const v8bf*)(r);
  u.h[1] = *(const v8bf*)(r + 8);
  return u.v;
}

// B fragment from TRANSPOSED weights [N][K]: 16 contiguous K per lane.
__device__ __forceinline__ v16bf load_bT_frag(const __bf16* baseT, int ldk, int lane) {
  int n = lane & 15, kb = (lane >> 4) << 4;
  return *(const v16bf*)(baseT + n * ldk + kb);
}

// Async global->LDS 16-byte copy (ASYNCcnt-tracked) + wait.
__device__ __forceinline__ void async_copy16(unsigned lds_off, const void* gptr) {
  asm volatile("global_load_async_to_lds_b128 %0, %1, off"
               :: "v"(lds_off), "v"(gptr) : "memory");
}
__device__ __forceinline__ void wait_async0() {
  asm volatile("s_wait_asynccnt 0x0" ::: "memory");
}

// ---------------------------------------------------------------------------
// Weight convert f32 -> bf16, TRANSPOSED + zero-padded: dst[pN][pK],
// dst[n][k] = src[(row0+k)*scols + col0+n]
// ---------------------------------------------------------------------------
__global__ __launch_bounds__(256) void cvtwT_kernel(
    const float* __restrict__ src, int scols, int row0, int col0,
    int rowsK, int colsN, __bf16* __restrict__ dst, int pK, int pN) {
  int idx = blockIdx.x * 256 + threadIdx.x;
  if (idx >= pN * pK) return;
  int n = idx / pK, k = idx - n * pK;
  float v = (k < rowsK && n < colsN) ? src[(size_t)(row0 + k) * scols + col0 + n] : 0.f;
  dst[idx] = (__bf16)v;
}

// ---------------------------------------------------------------------------
// Generic fused GEMM:  C = act( concat(A0[M][d0], A1[M][d1]) @ W + bias )
// W is bf16 TRANSPOSED [Nw][kp].  block=256 thr (8 waves), tile 32x64.
// grid = (Nw/64, M/32).  act: 0=none, 1=silu
// ---------------------------------------------------------------------------
__global__ __launch_bounds__(256) void gemm_kernel(
    const float* __restrict__ A0, int d0, const float* __restrict__ A1, int d1,
    const __bf16* __restrict__ Wt, int kp, const float* __restrict__ bias, int nb,
    float* __restrict__ C, int ldc, int M, int act) {
  __shared__ __bf16 sA[32 * 32];   // [row][k]
  __shared__ __bf16 sBt[64 * 32];  // [n][k]
  const int nbase = blockIdx.x * 64, rbase = blockIdx.y * 32;
  const int tid = threadIdx.x, lane = tid & 31, wv = tid >> 5;
  const int mrow = wv >> 2, ncol = wv & 3;
  const int K = d0 + d1, ksteps = (K + 31) >> 5;
  const bool a0_al = (d0 & 3) == 0, a1_al = ((d1 & 3) == 0) && ((d0 & 3) == 0);
  v8f acc = {};
  for (int k0 = 0; k0 < ksteps; ++k0) {
    if (k0 + 1 < ksteps)
      __builtin_prefetch(Wt + (size_t)nbase * kp + (k0 + 1) * 32, 0, 1);
    {  // stage A: thread -> 4 consecutive cols of one row; b128 fast path
      int r = tid >> 3, c0 = (tid & 7) << 2;
      int kg0 = k0 * 32 + c0;
      v4bf pk;
      if (a0_al && kg0 + 3 < d0) {
        float4 f = *(const float4*)(A0 + (size_t)(rbase + r) * d0 + kg0);
        pk[0] = (__bf16)f.x; pk[1] = (__bf16)f.y;
        pk[2] = (__bf16)f.z; pk[3] = (__bf16)f.w;
      } else if (a1_al && kg0 >= d0 && kg0 + 3 < K) {
        float4 f = *(const float4*)(A1 + (size_t)(rbase + r) * d1 + (kg0 - d0));
        pk[0] = (__bf16)f.x; pk[1] = (__bf16)f.y;
        pk[2] = (__bf16)f.z; pk[3] = (__bf16)f.w;
      } else {
#pragma unroll
        for (int t = 0; t < 4; ++t) {
          int kg = kg0 + t;
          float v = 0.f;
          if (kg < d0)     v = A0[(size_t)(rbase + r) * d0 + kg];
          else if (kg < K) v = A1[(size_t)(rbase + r) * d1 + (kg - d0)];
          pk[t] = (__bf16)v;
        }
      }
      *(v4bf*)(sA + r * 32 + c0) = pk;
    }
    {  // stage Wt tile [64][32]: one uint4 per thread
      int r = tid >> 2, c = tid & 3;
      *((uint4*)sBt + tid) =
          *(const uint4*)(Wt + (size_t)(nbase + r) * kp + k0 * 32 + c * 8);
    }
    __syncthreads();
    v16bf a = load_a_frag(sA + mrow * 16 * 32, 32, lane);
    v16bf b = load_bT_frag(sBt + (ncol * 16) * 32, 32, lane);
    acc = wmma_bf16(a, b, acc);
    __syncthreads();
  }
  const int n  = nbase + ncol * 16 + (lane & 15);
  const int mb = rbase + mrow * 16 + ((lane >> 4) << 3);
  float bv = (bias && n < nb) ? bias[n] : 0.f;
#pragma unroll
  for (int r = 0; r < 8; ++r) {
    float v = acc[r] + bv;
    if (act == 1) v = v * sigm(v);
    C[(size_t)(mb + r) * ldc + n] = v;
  }
}

// ---------------------------------------------------------------------------
// Fused edge kernel for one EGNN layer. grid=(N/16, B), block=256 (8 waves).
// Dynamic LDS = 311936 bytes.
// ---------------------------------------------------------------------------
__global__ __launch_bounds__(256) void edge_kernel(
    const float* __restrict__ Aterm,     // [B*N][256] (h@W1a + b1)
    const float* __restrict__ Bterm,     // [B*N][256] (h@W1b)
    const float* __restrict__ x_in,      // [B*N][3]
    const float* __restrict__ edge_mask, // [B][N][N]
    const float* __restrict__ w1c,       // [256] f32 (d2 row of edg1.w)
    const __bf16* __restrict__ W2t,      // [256n][256k] bf16 transposed
    const float* __restrict__ b2,        // [256]
    const float* __restrict__ waij, const float* __restrict__ baij,
    const __bf16* __restrict__ Wc1t,     // 2 blocks of [256n][256k] bf16
    const float* __restrict__ bc1,       // [512]
    const float* __restrict__ wc2,       // [512]
    const float* __restrict__ bc2,       // [1]
    float* __restrict__ agg_out,         // [B*N][256]
    float* __restrict__ x_out) {         // [B*N][3] = x + S/(2*(N-1))
  extern __shared__ char smem[];
  __bf16* sM   = (__bf16*)(smem);             // [128][256]  65536B
  __bf16* sMij = (__bf16*)(smem + 65536);     // [128][256]  65536B
  __bf16* sWt  = (__bf16*)(smem + 131072);    // [256][256] 131072B panel (transposed)
  float*  sAi  = (float*)(smem + 262144);     // [16][256]   16384B
  float*  sAgg = (float*)(smem + 278528);     // [16][256]   16384B
  float*  sBj  = (float*)(smem + 294912);     // [8][256]     8192B
  float*  sGate= (float*)(smem + 303104);     // [128]
  float*  sCw  = (float*)(smem + 303616);     // [128]
  float*  sXi  = (float*)(smem + 304128);     // [16][4]
  float*  sXj  = (float*)(smem + 304384);     // [8][4]
  float*  sXacc= (float*)(smem + 304512);     // [16][4]
  float*  sB2  = (float*)(smem + 304768);     // [256]
  float*  sWa  = (float*)(smem + 305792);     // [256]
  float*  sW1c = (float*)(smem + 306816);     // [256]
  float*  sBc1 = (float*)(smem + 307840);     // [512]
  float*  sWc2 = (float*)(smem + 309888);     // [512]

  const int b = blockIdx.y, i0 = blockIdx.x * 16;
  const int tid = threadIdx.x, lane = tid & 31, wv = tid >> 5;
  const int rowi = b * NN + i0;
  const int nl = lane & 15, mo = (lane >> 4) << 3;

  for (int idx = tid; idx < 16 * 256; idx += 256) {
    sAi[idx]  = Aterm[(size_t)(rowi + (idx >> 8)) * 256 + (idx & 255)];
    sAgg[idx] = 0.f;
  }
  sB2[tid] = b2[tid]; sWa[tid] = waij[tid]; sW1c[tid] = w1c[tid];
  for (int i = tid; i < 512; i += 256) { sBc1[i] = bc1[i]; sWc2[i] = wc2[i]; }
  if (tid < 48) {
    int ii = tid / 3, d = tid - ii * 3;
    sXi[ii * 4 + d]   = x_in[(size_t)(rowi + ii) * 3 + d];
    sXacc[ii * 4 + d] = 0.f;
  }
  __syncthreads();

  const unsigned sWt_off = (unsigned)(size_t)sWt;

  for (int j0 = 0; j0 < NN; j0 += 8) {
    // ---- stage Bterm rows, xj; async-stage W2 panel (128KB) ----
    for (int idx = tid; idx < 8 * 256; idx += 256)
      sBj[idx] = Bterm[(size_t)(b * NN + j0 + (idx >> 8)) * 256 + (idx & 255)];
    if (tid < 24) {
      int jj = tid / 3, d = tid - jj * 3;
      sXj[jj * 4 + d] = x_in[(size_t)(b * NN + j0 + jj) * 3 + d];
    }
    {
      const char* g = (const char*)W2t;
      __builtin_prefetch(Wc1t, 0, 1);
      for (int p = tid; p < 8192; p += 256)
        async_copy16(sWt_off + (unsigned)p * 16u, g + (size_t)p * 16);
      wait_async0();
    }
    __syncthreads();
    // ---- d2 per edge ----
    if (tid < 128) {
      int ii = tid >> 3, jj = tid & 7;
      float dx = sXi[ii * 4 + 0] - sXj[jj * 4 + 0];
      float dy = sXi[ii * 4 + 1] - sXj[jj * 4 + 1];
      float dz = sXi[ii * 4 + 2] - sXj[jj * 4 + 2];
      sGate[tid] = dx * dx + dy * dy + dz * dz;
    }
    __syncthreads();
    // ---- m1 = silu(Ai + Bj + d2*w1c)  -> sM (bf16, 8-wide packed) ----
    for (int g = tid; g < 128 * 32; g += 256) {
      int e = g >> 5, n0 = (g & 31) << 3;
      int ai = (e >> 3) * 256 + n0, bj = (e & 7) * 256 + n0;
      float d2 = sGate[e];
      v8bf out;
#pragma unroll
      for (int t = 0; t < 8; ++t) {
        float v = sAi[ai + t] + sBj[bj + t] + d2 * sW1c[n0 + t];
        v = v * sigm(v);
        out[t] = (__bf16)v;
      }
      *(v8bf*)(sM + e * 256 + n0) = out;
    }
    __syncthreads();
    // ---- edg2: [128][256] @ [256][256]^T WMMA, 2x2 register tiles ----
    for (int q = 0; q < 4; ++q) {
      int blk = q * 8 + wv;            // 0..31 : 4 Mblk x 8 Nblk
      int mt0 = (blk >> 3) * 2, nt0 = (blk & 7) * 2;
      v8f a00 = {}, a01 = {}, a10 = {}, a11 = {};
#pragma unroll
      for (int ks = 0; ks < 8; ++ks) {
        v16bf A0f = load_a_frag(sM + (mt0 * 16) * 256 + ks * 32, 256, lane);
        v16bf A1f = load_a_frag(sM + ((mt0 + 1) * 16) * 256 + ks * 32, 256, lane);
        v16bf B0f = load_bT_frag(sWt + (nt0 * 16) * 256 + ks * 32, 256, lane);
        v16bf B1f = load_bT_frag(sWt + ((nt0 + 1) * 16) * 256 + ks * 32, 256, lane);
        a00 = wmma_bf16(A0f, B0f, a00);
        a01 = wmma_bf16(A0f, B1f, a01);
        a10 = wmma_bf16(A1f, B0f, a10);
        a11 = wmma_bf16(A1f, B1f, a11);
      }
#pragma unroll
      for (int s = 0; s < 4; ++s) {
        v8f acc = (s == 0) ? a00 : (s == 1) ? a01 : (s == 2) ? a10 : a11;
        int mt = mt0 + (s >> 1), nt = nt0 + (s & 1);
        int n = nt * 16 + nl, mb = mt * 16 + mo;
        float bias = sB2[n];
#pragma unroll
        for (int r = 0; r < 8; ++r)
          sMij[(mb + r) * 256 + n] = (__bf16)(acc[r] + bias);
      }
    }
    __syncthreads();
    // ---- aij sigmoid gate * edge_mask (8-wide LDS reads) ----
    if (tid < 128) {
      int ii = tid >> 3, jj = tid & 7;
      float s = baij[0];
      const __bf16* row = sMij + tid * 256;
      for (int n = 0; n < 256; n += 8) {
        v8bf mv = *(const v8bf*)(row + n);
#pragma unroll
        for (int t = 0; t < 8; ++t) s += (float)mv[t] * sWa[n + t];
      }
      float m = edge_mask[((size_t)(b * NN) + i0 + ii) * NN + (j0 + jj)];
      sGate[tid] = sigm(s) * m;
      sCw[tid]   = bc2[0];
    }
    __syncthreads();
    for (int g = tid; g < 128 * 32; g += 256) {
      int e = g >> 5, n0 = (g & 31) << 3;
      float sc = sGate[e];
      v8bf mv = *(const v8bf*)(sMij + e * 256 + n0);
#pragma unroll
      for (int t = 0; t < 8; ++t) mv[t] = (__bf16)((float)mv[t] * sc);
      *(v8bf*)(sMij + e * 256 + n0) = mv;
    }
    __syncthreads();
    // ---- agg += sum_j mij (8-wide) ----
    for (int g = tid; g < 16 * 32; g += 256) {
      int ii = g >> 5, n0 = (g & 31) << 3;
      float acc8[8];
#pragma unroll
      for (int t = 0; t < 8; ++t) acc8[t] = sAgg[ii * 256 + n0 + t];
#pragma unroll
      for (int jj = 0; jj < 8; ++jj) {
        v8bf mv = *(const v8bf*)(sMij + (ii * 8 + jj) * 256 + n0);
#pragma unroll
        for (int t = 0; t < 8; ++t) acc8[t] += (float)mv[t];
      }
#pragma unroll
      for (int t = 0; t < 8; ++t) sAgg[ii * 256 + n0 + t] = acc8[t];
    }
    // ---- coord1 (two 256-col halves), fused tanh * wc2 contraction ----
    for (int half = 0; half < 2; ++half) {
      __syncthreads();
      {
        const char* g = (const char*)(Wc1t + half * 65536);
        for (int p = tid; p < 8192; p += 256)
          async_copy16(sWt_off + (unsigned)p * 16u, g + (size_t)p * 16);
        wait_async0();
      }
      __syncthreads();
      for (int q = 0; q < 4; ++q) {
        int blk = q * 8 + wv;
        int mt0 = (blk >> 3) * 2, nt0 = (blk & 7) * 2;
        v8f a00 = {}, a01 = {}, a10 = {}, a11 = {};
#pragma unroll
        for (int ks = 0; ks < 8; ++ks) {
          v16bf A0f = load_a_frag(sMij + (mt0 * 16) * 256 + ks * 32, 256, lane);
          v16bf A1f = load_a_frag(sMij + ((mt0 + 1) * 16) * 256 + ks * 32, 256, lane);
          v16bf B0f = load_bT_frag(sWt + (nt0 * 16) * 256 + ks * 32, 256, lane);
          v16bf B1f = load_bT_frag(sWt + ((nt0 + 1) * 16) * 256 + ks * 32, 256, lane);
          a00 = wmma_bf16(A0f, B0f, a00);
          a01 = wmma_bf16(A0f, B1f, a01);
          a10 = wmma_bf16(A1f, B0f, a10);
          a11 = wmma_bf16(A1f, B1f, a11);
        }
#pragma unroll
        for (int s = 0; s < 4; ++s) {
          v8f acc = (s == 0) ? a00 : (s == 1) ? a01 : (s == 2) ? a10 : a11;
          int mt = mt0 + (s >> 1), nt = nt0 + (s & 1);
          int o = half * 256 + nt * 16 + nl, mb = mt * 16 + mo;
          float bi = sBc1[o], wv2 = sWc2[o];
#pragma unroll
          for (int r = 0; r < 8; ++r)
            atomicAdd(&sCw[mb + r], fast_tanh(acc[r] + bi) * wv2);  // ds_add_f32
        }
      }
    }
    __syncthreads();
    // ---- x accumulation ----
    if (tid < 48) {
      int ii = tid / 3, d = tid - ii * 3;
      float s = 0.f;
#pragma unroll
      for (int jj = 0; jj < 8; ++jj)
        s += sCw[ii * 8 + jj] * (sXi[ii * 4 + d] - sXj[jj * 4 + d]);
      sXacc[ii * 4 + d] += s;
    }
    __syncthreads();
  }
  // ---- epilogue ----
  for (int idx = tid; idx < 16 * 256; idx += 256)
    agg_out[(size_t)(rowi + (idx >> 8)) * 256 + (idx & 255)] = sAgg[idx];
  if (tid < 48) {
    int ii = tid / 3, d = tid - ii * 3;
    x_out[(size_t)(rowi + ii) * 3 + d] =
        sXi[ii * 4 + d] + sXacc[ii * 4 + d] * (1.f / (2.f * (NN - 1)));
  }
}

// ---------------------------------------------------------------------------
__global__ __launch_bounds__(128) void hsum_kernel(const float* __restrict__ h9,
                                                   const float* __restrict__ nm,
                                                   float* __restrict__ hs) {
  int b = blockIdx.x, c = threadIdx.x;
  float s = 0.f;
  for (int n = 0; n < NN; ++n) s += h9[((size_t)b * NN + n) * 128 + c] * nm[b * NN + n];
  hs[b * 128 + c] = s;
}

__global__ __launch_bounds__(256) void pred_kernel(
    const float* __restrict__ hs,
    const float* __restrict__ w1, const float* __restrict__ b1,
    const float* __restrict__ w2, const float* __restrict__ b2,
    const float* __restrict__ w3, const float* __restrict__ b3,
    float* __restrict__ out) {
  __shared__ float s1[4 * 64];
  __shared__ float s2[4 * 32];
  const float al = 1.6732632423543772f, sc = 1.0507009873554805f;
  int tid = threadIdx.x;
  {
    int b = tid >> 6, o = tid & 63;
    float v = b1[o];
    for (int k = 0; k < 128; ++k) v += hs[b * 128 + k] * w1[k * 64 + o];
    s1[tid] = v > 0.f ? sc * v : sc * al * (__expf(v) - 1.f);
  }
  __syncthreads();
  if (tid < 128) {
    int b = tid >> 5, o = tid & 31;
    float v = b2[o];
    for (int k = 0; k < 64; ++k) v += s1[b * 64 + k] * w2[k * 32 + o];
    s2[tid] = v > 0.f ? sc * v : sc * al * (__expf(v) - 1.f);
  }
  __syncthreads();
  if (tid < 4) {
    float v = b3[0];
    for (int k = 0; k < 32; ++k) v += s2[tid * 32 + k] * w3[k];
    out[tid] = v;
  }
}

__global__ __launch_bounds__(256) void xcopy_kernel(const float* __restrict__ x9,
                                                    float* __restrict__ out) {
  int i = blockIdx.x * 256 + threadIdx.x;
  if (i < BB * NN * 3) out[4 + i] = x9[i];
}

// ---------------------------------------------------------------------------
// Host orchestration
// ---------------------------------------------------------------------------
static inline int pad32(int v) { return (v + 31) & ~31; }

extern "C" void kernel_launch(void* const* d_in, const int* in_sizes, int n_in,
                              void* d_out, int out_size, void* d_ws, size_t ws_size,
                              hipStream_t stream) {
  (void)in_sizes; (void)n_in; (void)out_size; (void)ws_size;
  const float* h_in = (const float*)d_in[0];
  const float* x_in = (const float*)d_in[1];
  const float* node_mask = (const float*)d_in[2];
  const float* edge_mask = (const float*)d_in[3];
  auto EMB = [&](int i, int wb) { return (const float*)d_in[4 + 2 * i + wb]; };
  // layer param order (sorted keys): aij, coord1, coord2, edg1, edg2, node1, node2
  auto LAY = [&](int l, int p, int wb) { return (const float*)d_in[22 + l * 14 + p * 2 + wb]; };
  auto PRD = [&](int i, int wb) { return (const float*)d_in[148 + 2 * i + wb]; };

  char* ws = (char*)d_ws;
  size_t off = 0;
  auto alloc = [&](size_t bytes) { size_t o = off; off = (off + bytes + 255) & ~(size_t)255; return o; };

  struct LWS { size_t w1a, w1b, w2, wc1, n1, n2, eW;
               int fea, feap, Kn, Knp, fo, fop, din, dinp; } L[9];
  for (int l = 0; l < 9; ++l) {
    L[l].fea = (l == 0) ? 6 : 128;
    L[l].feap = pad32(L[l].fea);
    L[l].Kn = L[l].fea + 256;            L[l].Knp = pad32(L[l].Kn);
    L[l].fo = 2 * L[l].Kn;               L[l].fop = (L[l].fo + 63) & ~63;   // 576 / 768
    L[l].din = L[l].fea + 128;           L[l].dinp = pad32(L[l].din);
    if (l == 6 || l == 7) { L[l].din = 256; L[l].dinp = 256; }
    L[l].w1a = alloc((size_t)L[l].feap * 256 * 2);
    L[l].w1b = alloc((size_t)L[l].feap * 256 * 2);
    L[l].w2  = alloc((size_t)256 * 256 * 2);
    L[l].wc1 = alloc((size_t)2 * 256 * 256 * 2);
    L[l].n1  = alloc((size_t)L[l].Knp * L[l].fop * 2);
    L[l].n2  = alloc((size_t)L[l].fop * 128 * 2);
    L[l].eW  = alloc((size_t)L[l].dinp * 128 * 2);
  }
  size_t oAt = alloc((size_t)1024 * 256 * 4);
  size_t oBt = alloc((size_t)1024 * 256 * 4);
  size_t oAgg = alloc((size_t)1024 * 256 * 4);
  size_t oNb = alloc((size_t)1024 * 768 * 4);
  size_t oEo = alloc((size_t)1024 * 128 * 4);
  size_t oHb[10];
  for (int l = 1; l <= 9; ++l) oHb[l] = alloc((size_t)1024 * 128 * 4);
  size_t oXa = alloc((size_t)BB * NN * 3 * 4);
  size_t oXb = alloc((size_t)BB * NN * 3 * 4);
  size_t oHs = alloc((size_t)BB * 128 * 4);

  auto BF = [&](size_t o) { return (__bf16*)(ws + o); };
  auto F  = [&](size_t o) { return (float*)(ws + o); };
  auto cvtT = [&](const float* src, int scols, int r0, int c0, int rowsK, int colsN,
                  __bf16* dst, int pK, int pN) {
    int n = pN * pK;
    hipLaunchKernelGGL(cvtwT_kernel, dim3((n + 255) / 256), dim3(256), 0, stream,
                       src, scols, r0, c0, rowsK, colsN, dst, pK, pN);
  };

  // -------- weight conversion (all transposed [N][K]) --------
  for (int l = 0; l < 9; ++l) {
    const float* e1w = LAY(l, 3, 0);
    cvtT(e1w, 256, 0, 0, L[l].fea, 256, BF(L[l].w1a), L[l].feap, 256);
    cvtT(e1w, 256, L[l].fea, 0, L[l].fea, 256, BF(L[l].w1b), L[l].feap, 256);
    cvtT(LAY(l, 4, 0), 256, 0, 0, 256, 256, BF(L[l].w2), 256, 256);
    cvtT(LAY(l, 1, 0), 512, 0, 0, 256, 256, BF(L[l].wc1), 256, 256);
    cvtT(LAY(l, 1, 0), 512, 0, 256, 256, 256, BF(L[l].wc1) + 65536, 256, 256);
    cvtT(LAY(l, 5, 0), L[l].fo, 0, 0, L[l].Kn, L[l].fo, BF(L[l].n1), L[l].Knp, L[l].fop);
    cvtT(LAY(l, 6, 0), 128, 0, 0, L[l].fo, 128, BF(L[l].n2), L[l].fop, 128);
    cvtT(EMB(l, 0), 128, 0, 0, L[l].din, 128, BF(L[l].eW), L[l].dinp, 128);
  }

  const size_t SMEM_EDGE = 311936;
  const float* xprev = x_in;
  float* xb[2] = { F(oXa), F(oXb) };

  for (int l = 0; l < 9; ++l) {
    const float* hcur = (l == 0) ? h_in : F(oHb[l]);
    float* xnext = xb[l & 1];
    hipLaunchKernelGGL(gemm_kernel, dim3(256 / 64, 32), dim3(256), 0, stream,
                       hcur, L[l].fea, (const float*)nullptr, 0,
                       BF(L[l].w1a), L[l].feap, LAY(l, 3, 1), 256, F(oAt), 256, 1024, 0);
    hipLaunchKernelGGL(gemm_kernel, dim3(256 / 64, 32), dim3(256), 0, stream,
                       hcur, L[l].fea, (const float*)nullptr, 0,
                       BF(L[l].w1b), L[l].feap, (const float*)nullptr, 0, F(oBt), 256, 1024, 0);
    hipLaunchKernelGGL(edge_kernel, dim3(NN / 16, BB), dim3(256), SMEM_EDGE, stream,
                       F(oAt), F(oBt), xprev, edge_mask,
                       LAY(l, 3, 0) + (size_t)2 * L[l].fea * 256,   // w1c row (f32)
                       BF(L[l].w2), LAY(l, 4, 1),
                       LAY(l, 0, 0), LAY(l, 0, 1),
                       BF(L[l].wc1), LAY(l, 1, 1),
                       LAY(l, 2, 0), LAY(l, 2, 1),
                       F(oAgg), xnext);
    hipLaunchKernelGGL(gemm_kernel, dim3(L[l].fop / 64, 32), dim3(256), 0, stream,
                       hcur, L[l].fea, F(oAgg), 256,
                       BF(L[l].n1), L[l].Knp, LAY(l, 5, 1), L[l].fo,
                       F(oNb), L[l].fop, 1024, 1);
    hipLaunchKernelGGL(gemm_kernel, dim3(128 / 64, 32), dim3(256), 0, stream,
                       F(oNb), L[l].fop, (const float*)nullptr, 0,
                       BF(L[l].n2), L[l].fop, LAY(l, 6, 1), 128, F(oEo), 128, 1024, 0);
    const float* ea = hcur; const float* eb = F(oEo);
    int da = L[l].fea, db = 128;
    if (l == 6) { ea = F(oHb[5]); eb = F(oHb[6]); da = 128; db = 128; }
    if (l == 7) { ea = F(oHb[6]); eb = F(oHb[7]); da = 128; db = 128; }
    hipLaunchKernelGGL(gemm_kernel, dim3(128 / 64, 32), dim3(256), 0, stream,
                       ea, da, eb, db, BF(L[l].eW), L[l].dinp, EMB(l, 1), 128,
                       F(oHb[l + 1]), 128, 1024, 0);
    xprev = xnext;
  }

  hipLaunchKernelGGL(hsum_kernel, dim3(BB), dim3(128), 0, stream,
                     F(oHb[9]), node_mask, F(oHs));
  hipLaunchKernelGGL(pred_kernel, dim3(1), dim3(256), 0, stream,
                     F(oHs), PRD(0, 0), PRD(0, 1), PRD(1, 0), PRD(1, 1),
                     PRD(2, 0), PRD(2, 1), (float*)d_out);
  hipLaunchKernelGGL(xcopy_kernel, dim3((BB * NN * 3 + 255) / 256), dim3(256), 0, stream,
                     xprev, (float*)d_out);
}
